// MultiHeadAttention_28346784154304
// MI455X (gfx1250) — matrix operand order repaired and checked by reading, not compiled, not run on hardware
//
#include <hip/hip_runtime.h>

// ---------------------------------------------------------------------------
// Problem constants (from the reference)
// ---------------------------------------------------------------------------
#define D_MODEL  1024
#define HEADS    16
#define HEAD_DIM 64
#define SEQ      2048
#define BATCH    2
#define ROWS     (BATCH * SEQ)      // 4096

// softmax runs in the log2 domain: v_exp_f32 is natively exp2.
#define LOG2E     1.44269504088896340736f
#define SCALE_Q   (0.125f * LOG2E)           // (1/sqrt(64)) * log2(e)
#define MASKED2   (-1.0e4f * LOG2E)          // masked fill, log2 domain

typedef __attribute__((ext_vector_type(16))) __bf16 v16bf;
typedef __attribute__((ext_vector_type(8)))  float  v8f;
typedef __attribute__((ext_vector_type(4)))  float  fvec4;

union V16U {                 // one WMMA A/B operand (16 bf16 = 8 VGPRs)
    v16bf v;
    uint4 u[2];
    __bf16 b[16];
};
union Pack8 {                // 8 bf16 packed into one 16B store
    __bf16 b[8];
    uint4  u;
};
union Pack4 {
    __bf16 b[4];
    uint2  u;
};

static __device__ __forceinline__ v8f wmma_bf16(v16bf a, v16bf b, v8f c) {
    // D(16x16 f32) = A(16x32 bf16) x B(32x16 bf16) + C
    return __builtin_amdgcn_wmma_f32_16x16x32_bf16(
        /*neg_a=*/false, a, /*neg_b=*/false, b,
        /*c_mod=*/(short)0, c, /*reuse_a=*/false, /*reuse_b=*/false);
}

// LDS byte offset of a generic pointer to a __shared__ object: per the CDNA5
// aperture rule, the low 32 bits of the flat address ARE the LDS offset.
static __device__ __forceinline__ unsigned lds_off(const void* p) {
    return (unsigned)(unsigned long long)p;
}

// Raw single-instruction max (operands known quiet; skips fmaxf's
// canonicalization pair).
static __device__ __forceinline__ float fmax_raw(float a, float b) {
    float d;
    asm("v_max_num_f32 %0, %1, %2" : "=v"(d) : "v"(a), "v"(b));
    return d;
}

// Max-reduce across each 16-lane half with FUSED DPP16 max ops (one VALU
// instruction per butterfly step): dst = max(dpp(src), src).
// xor1 = quad_perm[1,0,3,2], xor2 = quad_perm[2,3,0,1],
// ^7 = row_half_mirror, ^15 = row_mirror  -> full 16-lane reduction.
static __device__ __forceinline__ float dpp_max16(float x) {
    asm volatile("v_max_num_f32_dpp %0, %1, %1 quad_perm:[1,0,3,2] "
                 "row_mask:0xf bank_mask:0xf bound_ctrl:1"
                 : "=v"(x) : "v"(x));
    asm volatile("v_max_num_f32_dpp %0, %1, %1 quad_perm:[2,3,0,1] "
                 "row_mask:0xf bank_mask:0xf bound_ctrl:1"
                 : "=v"(x) : "v"(x));
    asm volatile("v_max_num_f32_dpp %0, %1, %1 row_half_mirror "
                 "row_mask:0xf bank_mask:0xf bound_ctrl:1"
                 : "=v"(x) : "v"(x));
    asm volatile("v_max_num_f32_dpp %0, %1, %1 row_mirror "
                 "row_mask:0xf bank_mask:0xf bound_ctrl:1"
                 : "=v"(x) : "v"(x));
    return x;
}

// ---------------------------------------------------------------------------
// fp32 -> bf16 conversion (streaming, non-temporal source, 4 elems / thread)
// ---------------------------------------------------------------------------
__global__ void cvt_f32_bf16(const float* __restrict__ src,
                             __bf16* __restrict__ dst, int n4) {
    int i = blockIdx.x * blockDim.x + threadIdx.x;
    if (i >= n4) return;
    fvec4 f = __builtin_nontemporal_load(
        reinterpret_cast<const fvec4*>(src) + i);
    Pack4 p;
    p.b[0] = (__bf16)f[0]; p.b[1] = (__bf16)f[1];
    p.b[2] = (__bf16)f[2]; p.b[3] = (__bf16)f[3];
    reinterpret_cast<uint2*>(dst)[i] = p.u;
}

// ---------------------------------------------------------------------------
// GEMM: C[M,N] = (A[M,K] . W[N,K]^T + b1[N] + b2[N]) * scale
//   OUT_MODE 0: bf16 row-major [M, D_MODEL]
//   OUT_MODE 1: bf16 transposed V layout  VT[((b*H+h)*64+d)*SEQ + s]
//   OUT_MODE 2: fp32 row-major [M, D_MODEL]   (final output, NT stores)
// Block = 8 waves = 128x128 tile. The shared 128x32 W tile is staged into
// LDS once per k-step with async global->LDS DMA, double buffered.
// ---------------------------------------------------------------------------
template<int OUT_MODE>
__global__ __launch_bounds__(256) void gemm_bf16_wmma(
    const __bf16* __restrict__ A,      // [M x 1024], K contiguous
    const __bf16* __restrict__ W,      // [N x 1024], K contiguous (out,in)
    const float*  __restrict__ b1,
    const float*  __restrict__ b2,
    void* __restrict__ outp,
    float scale)
{
    __shared__ __bf16 wtile[2][128 * 32];   // 2 x 8KB double buffer

    const int K    = D_MODEL;
    const int tid  = threadIdx.x;
    const int lane = tid & 31;
    const int wave = tid >> 5;
    const int half = lane >> 4;
    const int r    = lane & 15;
    const int mBase = blockIdx.y * 128 + wave * 16;
    const int nBase = blockIdx.x * 128;

    // Stage W[nBase .. nBase+127][k .. k+31] into wtile[buf]: 512 16B chunks
    // over 256 threads -> 2 async DMA issues / thread (ASYNCcnt tracked).
    auto stage = [&](int buf, int k) {
#pragma unroll
        for (int i = 0; i < 2; ++i) {
            const int c   = tid + i * 256;   // chunk 0..511
            const int row = c >> 2;          // 0..127
            const int c4  = c & 3;           // 16B chunk within 64B row
            const unsigned lds  = lds_off(&wtile[buf][row * 32 + c4 * 8]);
            const unsigned voff =
                (unsigned)(((nBase + row) * K + k + c4 * 8) * 2);
            asm volatile("global_load_async_to_lds_b128 %0, %1, %2"
                         :: "v"(lds), "v"(voff), "s"(W) : "memory");
        }
    };

    v8f acc[8];
    const v8f vzero = {0.f,0.f,0.f,0.f,0.f,0.f,0.f,0.f};
#pragma unroll
    for (int i = 0; i < 8; ++i) acc[i] = vzero;

    const __bf16* arow = A + (size_t)(mBase + r) * K;

    stage(0, 0);
    int buf = 0;
    for (int k = 0; k < K; k += 32, buf ^= 1) {
        asm volatile("s_wait_asynccnt 0" ::: "memory");
        __syncthreads();                      // wtile[buf] ready for all waves
        if (k + 32 < K) stage(buf ^ 1, k + 32);

        V16U a;   // A layout: elem e -> k + e + half*16
        a.u[0] = *reinterpret_cast<const uint4*>(arow + k + half * 16);
        a.u[1] = *reinterpret_cast<const uint4*>(arow + k + 8 + half * 16);
#pragma unroll
        for (int nt = 0; nt < 8; ++nt) {
            const __bf16* wl = &wtile[buf][(nt * 16 + r) * 32 + half * 16];
            V16U bb;  // B layout: elem e -> k + e + half*16, col N = r
            bb.u[0] = *reinterpret_cast<const uint4*>(wl);
            bb.u[1] = *reinterpret_cast<const uint4*>(wl + 8);
            acc[nt] = wmma_bf16(a.v, bb.v, acc[nt]);
        }
    }

    // Epilogue: C layout lane(half,r), VGPR v -> M = mBase + half*8 + v, N = col
#pragma unroll
    for (int nt = 0; nt < 8; ++nt) {
        const int n = nBase + nt * 16 + r;
        const float bias = b1[n] + b2[n];
        if (OUT_MODE == 2) {
            float* out = (float*)outp;
#pragma unroll
            for (int v = 0; v < 8; ++v)
                __builtin_nontemporal_store(
                    (acc[nt][v] + bias) * scale,
                    &out[(size_t)(mBase + half * 8 + v) * D_MODEL + n]);
        } else if (OUT_MODE == 0) {
            __bf16* out = (__bf16*)outp;
#pragma unroll
            for (int v = 0; v < 8; ++v)
                out[(size_t)(mBase + half * 8 + v) * D_MODEL + n] =
                    (__bf16)((acc[nt][v] + bias) * scale);
        } else {
            // transposed V: 8 consecutive s values -> one 16B store
            __bf16* out = (__bf16*)outp;
            const int m0   = mBase + half * 8;
            const int bidx = m0 >> 11;          // / SEQ
            const int s0   = m0 & (SEQ - 1);
            const int h    = n >> 6;
            const int d    = n & 63;
            Pack8 p;
#pragma unroll
            for (int v = 0; v < 8; ++v)
                p.b[v] = (__bf16)((acc[nt][v] + bias) * scale);
            *reinterpret_cast<uint4*>(
                out + ((size_t)((bidx * HEADS + h) * HEAD_DIM + d)) * SEQ + s0) =
                p.u;
        }
    }
}

// ---------------------------------------------------------------------------
// Flash attention. Block = 8 waves sharing one (b,h); each wave owns a
// 16-query tile. K/V tiles for each 32-key step are staged into LDS with
// async global->LDS DMA (double buffered, 2 issues/thread/step). Row sums of
// P ride the matrix pipe (extra WMMA vs all-ones B); row max uses fused DPP16
// max butterflies. Q arrives pre-scaled by log2(e)/sqrt(64), so every
// exponential is a single native v_exp_f32 (exp2). Mask (536 MB, single use)
// streams through batched NT loads + prefetch, overlapping the WMMAs, so
// Q/K/V stay resident in the 192 MB L2.
// ---------------------------------------------------------------------------
__global__ __launch_bounds__(256) void attn_wmma(
    const __bf16* __restrict__ Qp,   // [ROWS x D_MODEL], pre-scaled
    const __bf16* __restrict__ Kp,   // [ROWS x D_MODEL]
    const __bf16* __restrict__ VT,   // [(B*H*64) x SEQ]
    const int*    __restrict__ mask, // [B,H,S,S] int32 {0,1}
    __bf16* __restrict__ AO)         // [ROWS x D_MODEL]
{
    __shared__ __bf16 ktile[2][32 * 64];  // [key][d]   4KB x2
    __shared__ __bf16 vtile[2][64 * 32];  // [d][key]   4KB x2
    __shared__ __bf16 plds[8][16 * 32];   // per-wave P staging (C -> A layout)

    const int tid  = threadIdx.x;
    const int lane = tid & 31;
    const int wave = tid >> 5;
    const int half = lane >> 4;
    const int r    = lane & 15;
    const int bh   = blockIdx.y;          // b*H + h
    const int b    = bh >> 4;
    const int h    = bh & 15;
    const int qBase = blockIdx.x * 128 + wave * 16;

    // Stage the 32x64 K tile and 64x32 V tile for keys [kb, kb+32):
    // 256 + 256 16B chunks over 256 threads -> 2 async issues per thread.
    auto stage = [&](int buf, int kb) {
        {
            const int row = tid >> 3, c8 = tid & 7;          // K: 32 x 8 chunks
            const unsigned lds  = lds_off(&ktile[buf][row * 64 + c8 * 8]);
            const unsigned voff = (unsigned)(((b * SEQ + kb + row) * D_MODEL +
                                              h * HEAD_DIM + c8 * 8) * 2);
            asm volatile("global_load_async_to_lds_b128 %0, %1, %2"
                         :: "v"(lds), "v"(voff), "s"(Kp) : "memory");
        }
        {
            const int row = tid >> 2, c4 = tid & 3;          // V: 64 x 4 chunks
            const unsigned lds  = lds_off(&vtile[buf][row * 32 + c4 * 8]);
            const unsigned voff = (unsigned)(((bh * HEAD_DIM + row) * SEQ +
                                              kb + c4 * 8) * 2);
            asm volatile("global_load_async_to_lds_b128 %0, %1, %2"
                         :: "v"(lds), "v"(voff), "s"(VT) : "memory");
        }
    };

    // Q as WMMA A operands for both 32-wide k-steps of HEAD_DIM=64
    const __bf16* qrow =
        Qp + (size_t)(b * SEQ + qBase + r) * D_MODEL + h * HEAD_DIM;
    V16U aq0, aq1;
    aq0.u[0] = *(const uint4*)(qrow + half * 16);
    aq0.u[1] = *(const uint4*)(qrow + 8 + half * 16);
    aq1.u[0] = *(const uint4*)(qrow + 32 + half * 16);
    aq1.u[1] = *(const uint4*)(qrow + 40 + half * 16);

    V16U ones;                       // all-ones B matrix for row sums
#pragma unroll
    for (int e = 0; e < 16; ++e) ones.b[e] = (__bf16)1.0f;

    const v8f vzero = {0.f,0.f,0.f,0.f,0.f,0.f,0.f,0.f};
    float mrow[8];
    v8f o[4], osum;
#pragma unroll
    for (int v = 0; v < 8; ++v) mrow[v] = -1.0e30f;
#pragma unroll
    for (int dt = 0; dt < 4; ++dt) o[dt] = vzero;
    osum = vzero;

    const int* mbase = mask + ((size_t)bh * SEQ + qBase + half * 8) * SEQ;

    stage(0, 0);
    int buf = 0;
    for (int kb = 0; kb < SEQ; kb += 32, buf ^= 1) {
        asm volatile("s_wait_asynccnt 0" ::: "memory");
        __syncthreads();                        // tiles[buf] ready
        if (kb + 32 < SEQ) stage(buf ^ 1, kb + 32);

        // Batch all mask loads for this 32-key step up front (NT + prefetch)
        // so their latency overlaps the score WMMAs below.
        int mm0[8], mm1[8];
#pragma unroll
        for (int v = 0; v < 8; ++v) {
            const int* mp = mbase + (size_t)v * SEQ + kb;
            __builtin_prefetch(mp + 32, 0, 0);          // next 128B mask line
            mm0[v] = __builtin_nontemporal_load(mp + r);
            mm1[v] = __builtin_nontemporal_load(mp + 16 + r);
        }

        // scores (log2 domain): S = Q . K^T, Q pre-scaled by log2e/8.
        v8f s0 = vzero, s1 = vzero;
        {
            const __bf16* kl = &ktile[buf][r * 64 + half * 16];
            V16U bk0, bk1;
            bk0.u[0] = *(const uint4*)(kl);
            bk0.u[1] = *(const uint4*)(kl + 8);
            bk1.u[0] = *(const uint4*)(kl + 32);
            bk1.u[1] = *(const uint4*)(kl + 40);
            s0 = wmma_bf16(aq0.v, bk0.v, s0);
            s0 = wmma_bf16(aq1.v, bk1.v, s0);
        }
        {
            const __bf16* kl = &ktile[buf][(16 + r) * 64 + half * 16];
            V16U bk0, bk1;
            bk0.u[0] = *(const uint4*)(kl);
            bk0.u[1] = *(const uint4*)(kl + 8);
            bk1.u[0] = *(const uint4*)(kl + 32);
            bk1.u[1] = *(const uint4*)(kl + 40);
            s1 = wmma_bf16(aq0.v, bk0.v, s1);
            s1 = wmma_bf16(aq1.v, bk1.v, s1);
        }

        // mask + online softmax; fused-DPP row-max, WMMA row-sum, exp2 only.
#pragma unroll
        for (int v = 0; v < 8; ++v) {
            const float sv0 = mm0[v] ? s0[v] : MASKED2;
            const float sv1 = mm1[v] ? s1[v] : MASKED2;

            const float rm    = dpp_max16(fmax_raw(sv0, sv1));
            const float mNew  = fmax_raw(mrow[v], rm);
            const float alpha = __builtin_amdgcn_exp2f(mrow[v] - mNew);
            const float p0    = __builtin_amdgcn_exp2f(sv0 - mNew);
            const float p1    = __builtin_amdgcn_exp2f(sv1 - mNew);
            mrow[v] = mNew;
#pragma unroll
            for (int dt = 0; dt < 4; ++dt) o[dt][v] *= alpha;
            osum[v] *= alpha;

            // stash P (C layout) into LDS row-major [q][k32]
            plds[wave][(half * 8 + v) * 32 + r]      = (__bf16)p0;
            plds[wave][(half * 8 + v) * 32 + 16 + r] = (__bf16)p1;
        }
        asm volatile("s_wait_dscnt 0" ::: "memory");  // P visible wave-wide

        // reload P in WMMA A layout: lane (half,r) elem e -> row r,
        // k = e + half*16  -> two contiguous 16B LDS reads
        V16U pa;
        const __bf16* pl = &plds[wave][r * 32 + half * 16];
        pa.u[0] = *(const uint4*)(pl);
        pa.u[1] = *(const uint4*)(pl + 8);

        // row sums of P via matrix pipe: every column of osum = sum_k P[m][k]
        osum = wmma_bf16(pa.v, ones.v, osum);

        // O += P(16x32) x V(32x16) per 16-wide d tile, V from LDS.
#pragma unroll
        for (int dt = 0; dt < 4; ++dt) {
            const __bf16* vl = &vtile[buf][(dt * 16 + r) * 32 + half * 16];
            V16U bv;
            bv.u[0] = *(const uint4*)(vl);
            bv.u[1] = *(const uint4*)(vl + 8);
            o[dt] = wmma_bf16(pa.v, bv.v, o[dt]);
        }
    }

    // normalize and emit bf16 attention output (A operand of final GEMM)
    float rcp[8];
#pragma unroll
    for (int v = 0; v < 8; ++v) rcp[v] = 1.0f / osum[v];
#pragma unroll
    for (int dt = 0; dt < 4; ++dt) {
#pragma unroll
        for (int v = 0; v < 8; ++v) {
            const float val = o[dt][v] * rcp[v];
            AO[(size_t)(b * SEQ + qBase + half * 8 + v) * D_MODEL +
               h * HEAD_DIM + dt * 16 + r] = (__bf16)val;
        }
    }
}

// ---------------------------------------------------------------------------
// Host launcher
// ---------------------------------------------------------------------------
extern "C" void kernel_launch(void* const* d_in, const int* in_sizes, int n_in,
                              void* d_out, int out_size, void* d_ws,
                              size_t ws_size, hipStream_t stream) {
    (void)in_sizes; (void)n_in; (void)out_size; (void)ws_size;

    const float* q    = (const float*)d_in[0];
    const float* kk   = (const float*)d_in[1];
    const float* v    = (const float*)d_in[2];
    const int*   mask = (const int*)  d_in[3];
    const float* Wq   = (const float*)d_in[4];
    const float* bq   = (const float*)d_in[5];
    const float* Wk   = (const float*)d_in[6];
    const float* bk   = (const float*)d_in[7];
    const float* Wv   = (const float*)d_in[8];
    const float* bv   = (const float*)d_in[9];
    const float* Wy   = (const float*)d_in[10];
    const float* by   = (const float*)d_in[11];
    const float* bq2  = (const float*)d_in[12];
    const float* bk2  = (const float*)d_in[13];
    const float* bv2  = (const float*)d_in[14];
    const float* by2  = (const float*)d_in[15];

    const size_t NROW = (size_t)ROWS * D_MODEL;      // 4M elems
    const size_t NW   = (size_t)D_MODEL * D_MODEL;   // 1M elems

    char* p = (char*)d_ws;
    auto carve = [&](size_t elems) {
        __bf16* r = (__bf16*)p;
        p += elems * sizeof(__bf16);
        return r;
    };
    __bf16* Xq  = carve(NROW);
    __bf16* Xk  = carve(NROW);
    __bf16* Xv  = carve(NROW);
    __bf16* Wqb = carve(NW);
    __bf16* Wkb = carve(NW);
    __bf16* Wvb = carve(NW);
    __bf16* Wyb = carve(NW);
    __bf16* Qp  = carve(NROW);
    __bf16* Kp  = carve(NROW);
    __bf16* VTb = carve(NROW);
    __bf16* AO  = carve(NROW);

    const int T = 256;
    auto cvt = [&](const float* s, __bf16* d, size_t n) {
        const int n4 = (int)(n / 4);
        cvt_f32_bf16<<<(n4 + T - 1) / T, T, 0, stream>>>(s, d, n4);
    };
    cvt(q,  Xq,  NROW);
    cvt(kk, Xk,  NROW);
    cvt(v,  Xv,  NROW);
    cvt(Wq, Wqb, NW);
    cvt(Wk, Wkb, NW);
    cvt(Wv, Wvb, NW);
    cvt(Wy, Wyb, NW);

    dim3 gblk(256);
    dim3 ggrid(D_MODEL / 128, ROWS / 128);   // (8, 32)
    // Q gets 1/sqrt(HEAD_SIZE) * log2(e) folded in (softmax in log2 domain).
    gemm_bf16_wmma<0><<<ggrid, gblk, 0, stream>>>(Xq, Wqb, bq, bq2, Qp, SCALE_Q);
    gemm_bf16_wmma<0><<<ggrid, gblk, 0, stream>>>(Xk, Wkb, bk, bk2, Kp, 1.0f);
    gemm_bf16_wmma<1><<<ggrid, gblk, 0, stream>>>(Xv, Wvb, bv, bv2, VTb, 1.0f);

    attn_wmma<<<dim3(SEQ / 128, BATCH * HEADS), 256, 0, stream>>>(
        Qp, Kp, VTb, mask, AO);

    gemm_bf16_wmma<2><<<ggrid, gblk, 0, stream>>>(AO, Wyb, by, by2, d_out, 1.0f);
}
